// ParallelRiskAggregationNN_35227321762459
// MI455X (gfx1250) — matrix operand order repaired and checked by reading, not compiled, not run on hardware
//
#include <hip/hip_runtime.h>

typedef __attribute__((ext_vector_type(16))) _Float16 v16h;
typedef __attribute__((ext_vector_type(8)))  _Float16 v8h;
typedef __attribute__((ext_vector_type(8)))  float    v8f;
typedef __attribute__((ext_vector_type(4)))  float    v4f;

#define NB   32768
#define NDIM 64
#define WAVES 8
#define ROWS_PER_WAVE 32
#define ROWS_PER_TILE 256            // 8 waves * 32 rows
#define H_SPLITS 16                  // batch splits for h kernel
#define H_TILES (NB / (H_SPLITS * ROWS_PER_TILE))   // 8 tiles per block

__device__ __forceinline__ v16h cat8(v8h lo, v8h hi) {
    return __builtin_shufflevector(lo, hi, 0,1,2,3,4,5,6,7,8,9,10,11,12,13,14,15);
}

// Load the wave's A-side (weight) fragment for M-tile mot, K-half kt from
// wT[k_out*64 + k_in] in LDS. Lane holds row M = 16*mot + (lane&15),
// elements e0-7 -> K = 32*kt + 8*g + e, e8-15 -> K = 32*kt + 16 + 8*g + (e-8).
__device__ __forceinline__ v16h load_a_frag(const _Float16* __restrict__ wT,
                                            int mot, int kt, int m, int g) {
    const _Float16* arow = wT + (16 * mot + m) * 64 + 32 * kt;
    return cat8(*(const v8h*)(arow + 8 * g), *(const v8h*)(arow + 16 + 8 * g));
}

// One MLP layer for one wave, 32 batch rows, weights resident in registers:
//   act[n][k] (32x64 f16, wave-private LDS)  <-  relu(act @ W + bias)
// A = W^T tiles (M = k_out), B = act tiles (N = batch row).
__device__ __forceinline__ void wave_mlp_layer_reg(_Float16* __restrict__ act,
                                                   const v16h (*__restrict__ aw)[2],
                                                   const float* __restrict__ bias,
                                                   int lane) {
    const int m = lane & 15;
    const int g = lane >> 4;

    // B fragments (activations): lane holds N = 16*nt + m, e0-7 -> K = 32*kt + 8*g + e, e8-15 -> K = 32*kt + 16 + 8*g + (e-8).
    v16h b[2][2];
    #pragma unroll
    for (int nt = 0; nt < 2; ++nt) {
        const _Float16* brow = act + (16 * nt + m) * 64 + 8 * g;
        #pragma unroll
        for (int kt = 0; kt < 2; ++kt) {
            const v8h lo = *(const v8h*)(brow + 32 * kt);
            const v8h hi = *(const v8h*)(brow + 32 * kt + 16);
            b[nt][kt] = cat8(lo, hi);
        }
    }

    v8f acc[4][2];
    #pragma unroll
    for (int mot = 0; mot < 4; ++mot) {
        #pragma unroll
        for (int nt = 0; nt < 2; ++nt) {
            v8f c = {};
            c = __builtin_amdgcn_wmma_f32_16x16x32_f16(false, aw[mot][0], false, b[nt][0], (short)0, c, false, false);
            c = __builtin_amdgcn_wmma_f32_16x16x32_f16(false, aw[mot][1], false, b[nt][1], (short)0, c, false, false);
            acc[mot][nt] = c;
        }
    }

    // D layout: lane holds batch row n = 16*nt + m, k_out = 16*mot + 8*g + r
    // -> 8 contiguous k_out per lane: bias+relu, pack, single b128 store.
    #pragma unroll
    for (int mot = 0; mot < 4; ++mot) {
        const v4f blo = *(const v4f*)(bias + 16 * mot + 8 * g);
        const v4f bhi = *(const v4f*)(bias + 16 * mot + 8 * g + 4);
        #pragma unroll
        for (int nt = 0; nt < 2; ++nt) {
            v8h o;
            #pragma unroll
            for (int r = 0; r < 8; ++r) {
                float v = acc[mot][nt][r] + (r < 4 ? blo[r] : bhi[r - 4]);
                v = v > 0.f ? v : 0.f;
                o[r] = (_Float16)v;
            }
            *(v8h*)(act + (16 * nt + m) * 64 + 16 * mot + 8 * g) = o;
        }
    }
}

// h-path: grid (H_SPLITS, 64); persistent over H_TILES batch tiles.
__global__ __launch_bounds__(256) void h_path_kernel(
        const float* __restrict__ x, const float* __restrict__ h_w1,
        const float* __restrict__ h_b1, const float* __restrict__ h_wh,
        const float* __restrict__ h_bh, const float* __restrict__ h_wo,
        const float* __restrict__ h_bo, float* __restrict__ out) {
    __shared__ __align__(16) _Float16 sW[2][64 * 64];      // W^T per layer, f16
    __shared__ __align__(16) _Float16 sAct[WAVES][ROWS_PER_WAVE * 64];
    __shared__ __align__(16) float sB[2][64];
    __shared__ __align__(16) float sW1[64], sB1[64], sWo[64];

    const int tid = threadIdx.x;
    const int d = blockIdx.y;

    // One-time per block: load + transpose per-head 64x64 weights into LDS f16.
    #pragma unroll
    for (int l = 0; l < 2; ++l) {
        const float* w = h_wh + ((size_t)l * NDIM + d) * 64 * 64;
        for (int idx = tid; idx < 64 * 64; idx += 256) {
            const int hh = idx >> 6, k = idx & 63;
            sW[l][k * 64 + hh] = (_Float16)w[idx];     // sW[k_out][k_in]
        }
    }
    if (tid < 64) {
        sW1[tid]   = h_w1[d * 64 + tid];
        sB1[tid]   = h_b1[d * 64 + tid];
        sWo[tid]   = h_wo[d * 64 + tid];
        sB[0][tid] = h_bh[((size_t)0 * NDIM + d) * 64 + tid];
        sB[1][tid] = h_bh[((size_t)1 * NDIM + d) * 64 + tid];
    }
    __syncthreads();

    const int wave = tid >> 5;
    const int lane = tid & 31;
    const int m = lane & 15;
    const int g = lane >> 4;
    _Float16* act = &sAct[wave][0];
    const int n = lane;                      // each lane owns one batch row
    _Float16* arow = act + n * 64;
    const float bo = h_bo[d];

    // Hoist all weight A-fragments into registers (2 layers x 4 M-tiles x 2 K).
    v16h aw[2][4][2];
    #pragma unroll
    for (int l = 0; l < 2; ++l)
        #pragma unroll
        for (int mot = 0; mot < 4; ++mot)
            #pragma unroll
            for (int kt = 0; kt < 2; ++kt)
                aw[l][mot][kt] = load_a_frag(sW[l], mot, kt, m, g);

    const int blockRow0 = blockIdx.x * (NB / H_SPLITS);

    for (int it = 0; it < H_TILES; ++it) {
        const int rowBase = blockRow0 + it * ROWS_PER_TILE + wave * ROWS_PER_WAVE;
        const int row = rowBase + n;

        // init: act[n][k] = relu(second[row]*w1[k] + b1[k])
        const float sec = x[(size_t)row * 128 + 64 + d];
        #pragma unroll
        for (int c = 0; c < 8; ++c) {
            const v4f w0 = *(const v4f*)(sW1 + c * 8);
            const v4f w1v = *(const v4f*)(sW1 + c * 8 + 4);
            const v4f b0 = *(const v4f*)(sB1 + c * 8);
            const v4f b1v = *(const v4f*)(sB1 + c * 8 + 4);
            v8h o;
            #pragma unroll
            for (int i = 0; i < 4; ++i) {
                float v = sec * w0[i] + b0[i];   o[i]     = (_Float16)(v > 0.f ? v : 0.f);
            }
            #pragma unroll
            for (int i = 0; i < 4; ++i) {
                float v = sec * w1v[i] + b1v[i]; o[i + 4] = (_Float16)(v > 0.f ? v : 0.f);
            }
            *(v8h*)(arow + c * 8) = o;
        }
        // activations are wave-private; DS ops of one wave are in-order.
        wave_mlp_layer_reg(act, aw[0], sB[0], lane);
        wave_mlp_layer_reg(act, aw[1], sB[1], lane);

        // h_out[row][d] = act[row] . wo + bo
        float s = 0.f;
        #pragma unroll
        for (int c = 0; c < 8; ++c) {
            const v8h av = *(const v8h*)(arow + c * 8);
            const v4f w0 = *(const v4f*)(sWo + c * 8);
            const v4f w1v = *(const v4f*)(sWo + c * 8 + 4);
            #pragma unroll
            for (int i = 0; i < 4; ++i) s += (float)av[i] * w0[i];
            #pragma unroll
            for (int i = 0; i < 4; ++i) s += (float)av[i + 4] * w1v[i];
        }
        out[(size_t)row * 64 + d] = s + bo;
    }
}

// g-path: grid NB/256 blocks. 3 GEMM layers + final dot (small: ~2% of FLOPs).
__global__ __launch_bounds__(256) void g_path_kernel(
        const float* __restrict__ x, const float* __restrict__ g_w1,
        const float* __restrict__ g_b1, const float* __restrict__ g_wh,
        const float* __restrict__ g_bh, const float* __restrict__ g_wo,
        const float* __restrict__ g_bo, float* __restrict__ out) {
    __shared__ __align__(16) _Float16 sW[3][64 * 64];
    __shared__ __align__(16) _Float16 sAct[WAVES][ROWS_PER_WAVE * 64];
    __shared__ __align__(16) float sB[3][64];
    __shared__ __align__(16) float sWo[64];

    const int tid = threadIdx.x;
    for (int idx = tid; idx < 64 * 64; idx += 256) {
        const int hh = idx >> 6, k = idx & 63;
        sW[0][k * 64 + hh] = (_Float16)g_w1[idx];
        sW[1][k * 64 + hh] = (_Float16)g_wh[idx];
        sW[2][k * 64 + hh] = (_Float16)g_wh[64 * 64 + idx];
    }
    if (tid < 64) {
        sB[0][tid] = g_b1[tid];
        sB[1][tid] = g_bh[tid];
        sB[2][tid] = g_bh[64 + tid];
        sWo[tid]   = g_wo[tid];
    }
    __syncthreads();

    const int wave = tid >> 5;
    const int lane = tid & 31;
    const int m = lane & 15;
    const int g = lane >> 4;
    _Float16* act = &sAct[wave][0];
    const int rowBase = blockIdx.x * ROWS_PER_TILE + wave * ROWS_PER_WAVE;
    const int n = lane;
    _Float16* arow = act + n * 64;

    v16h aw[3][4][2];
    #pragma unroll
    for (int l = 0; l < 3; ++l)
        #pragma unroll
        for (int mot = 0; mot < 4; ++mot)
            #pragma unroll
            for (int kt = 0; kt < 2; ++kt)
                aw[l][mot][kt] = load_a_frag(sW[l], mot, kt, m, g);

    // stage 'first' row of this lane as f16
    const float* xrow = x + (size_t)(rowBase + n) * 128;
    #pragma unroll
    for (int c = 0; c < 8; ++c) {
        const v4f x0 = *(const v4f*)(xrow + c * 8);
        const v4f x1 = *(const v4f*)(xrow + c * 8 + 4);
        v8h o;
        #pragma unroll
        for (int i = 0; i < 4; ++i) o[i]     = (_Float16)x0[i];
        #pragma unroll
        for (int i = 0; i < 4; ++i) o[i + 4] = (_Float16)x1[i];
        *(v8h*)(arow + c * 8) = o;
    }
    wave_mlp_layer_reg(act, aw[0], sB[0], lane);
    wave_mlp_layer_reg(act, aw[1], sB[1], lane);
    wave_mlp_layer_reg(act, aw[2], sB[2], lane);

    float s = 0.f;
    #pragma unroll
    for (int c = 0; c < 8; ++c) {
        const v8h av = *(const v8h*)(arow + c * 8);
        const v4f w0 = *(const v4f*)(sWo + c * 8);
        const v4f w1v = *(const v4f*)(sWo + c * 8 + 4);
        #pragma unroll
        for (int i = 0; i < 4; ++i) s += (float)av[i] * w0[i];
        #pragma unroll
        for (int i = 0; i < 4; ++i) s += (float)av[i + 4] * w1v[i];
    }
    out[(size_t)NB * 64 + (rowBase + n)] = s + g_bo[0];
}

extern "C" void kernel_launch(void* const* d_in, const int* in_sizes, int n_in,
                              void* d_out, int out_size, void* d_ws, size_t ws_size,
                              hipStream_t stream) {
    (void)in_sizes; (void)n_in; (void)out_size; (void)d_ws; (void)ws_size;
    const float* x    = (const float*)d_in[0];
    const float* h_w1 = (const float*)d_in[1];
    const float* h_b1 = (const float*)d_in[2];
    const float* h_wh = (const float*)d_in[3];
    const float* h_bh = (const float*)d_in[4];
    const float* h_wo = (const float*)d_in[5];
    const float* h_bo = (const float*)d_in[6];
    const float* g_w1 = (const float*)d_in[7];
    const float* g_b1 = (const float*)d_in[8];
    const float* g_wh = (const float*)d_in[9];
    const float* g_bh = (const float*)d_in[10];
    const float* g_wo = (const float*)d_in[11];
    const float* g_bo = (const float*)d_in[12];
    float* out = (float*)d_out;

    dim3 hgrid(H_SPLITS, NDIM);   // 16 x 64 = 1024 persistent blocks
    h_path_kernel<<<hgrid, 256, 0, stream>>>(x, h_w1, h_b1, h_wh, h_bh, h_wo, h_bo, out);
    g_path_kernel<<<NB / ROWS_PER_TILE, 256, 0, stream>>>(x, g_w1, g_b1, g_wh, g_bh, g_wo, g_bo, out);
}
